// AutoformerAttention_45243185496309
// MI455X (gfx1250) — matrix operand order repaired and checked by reading, not compiled, not run on hardware
//
#include <hip/hip_runtime.h>
#include <stdint.h>

// ---------------- problem constants ----------------
#define T_SEQ   4096
#define EMB     1024
#define HEADS   16
#define HDIM    64
#define BATCH   8
#define BNHEADS 128          // BATCH * HEADS
#define MROWS   32768        // BATCH * T_SEQ
#define TOPK    24           // floor(3 * ln(4096))

typedef __attribute__((ext_vector_type(16))) _Float16     v16h;
typedef __attribute__((ext_vector_type(8)))  float        v8f;
typedef __attribute__((ext_vector_type(4)))  unsigned int v4u;
typedef __attribute__((ext_vector_type(8)))  unsigned int v8u;

// ---------------- WMMA fragment loaders (gfx1250 layouts) ----------------
// A matrix 16x32 f16 (ISA 7.12.2): lanes 0-15 -> M=0..15, v0..v3 hold K={0..7},
// v4..v7 hold K={16..23}; lanes 16-31 add +8 to the K base of each half.
__device__ __forceinline__ v16h load_a_frag(const _Float16* __restrict__ s,
                                            int lda, int row0, int col0) {
    const int lane = threadIdx.x & 31;
    const _Float16* r = s + (size_t)(row0 + (lane & 15)) * lda + col0;
    const int hi = (lane >> 4) << 3;   // 0 or 8
    v16h a;
#pragma unroll
    for (int v = 0; v < 8; ++v) {
        const int kb = ((v < 4) ? (2 * v) : (16 + 2 * (v - 4))) + hi;
        a[2 * v]     = r[kb];
        a[2 * v + 1] = r[kb + 1];
    }
    return a;
}

// B matrix 32x16 f16: N = lane%16, K = (lane/16)*8 + e for e<8, 16 + (lane/16)*8 + (e-8) for e>=8.
// LDS tile is stored [n][k] (k contiguous); same K order as the A fragment.
__device__ __forceinline__ v16h load_b_frag(const _Float16* __restrict__ s,
                                            int ldb, int n0, int k0) {
    const int lane = threadIdx.x & 31;
    const _Float16* r = s + (size_t)(n0 + (lane & 15)) * ldb + k0 + ((lane >> 4) << 3);
    v16h b;
#pragma unroll
    for (int e = 0; e < 16; ++e) b[e] = r[(e < 8) ? e : (e + 8)];
    return b;
}

// ---------------- TDM: 1-D contiguous global -> LDS copy --------------------
// Builds a Tensor DMA Descriptor (CDNA5 ISA):
//   group0: count=1 | lds_addr | global_addr[56:0] | type=2
//   group1: data_size=1 (2B), tensor_dim0=tile_dim0=nelem, dim1=1, stride=nelem
// Issued by one wave; completion via TENSORcnt.
__device__ __forceinline__ void tdm_load_f16(unsigned lds_byte_off,
                                             const _Float16* gptr,
                                             unsigned nelem) {
    const unsigned long long ga = (unsigned long long)(uintptr_t)gptr;
    v4u g0;
    g0[0] = 1u;                                                  // count=1
    g0[1] = lds_byte_off;                                        // D#.lds_addr
    g0[2] = (unsigned)(ga & 0xFFFFFFFFu);                        // gaddr[31:0]
    g0[3] = (unsigned)((ga >> 32) & 0x01FFFFFFu) | (2u << 30);   // gaddr hi | type=2
    v8u g1;
    g1[0] = 1u << 16;                            // data_size = 1 (2 bytes/elem)
    g1[1] = (nelem & 0xFFFFu) << 16;             // tensor_dim0[15:0]
    g1[2] = (nelem >> 16) | (1u << 16);          // tensor_dim0[31:16] | tensor_dim1=1
    g1[3] = (nelem & 0xFFFFu) << 16;             // tile_dim0 = nelem
    g1[4] = 0u;                                  // tile_dim1/2 unused
    g1[5] = nelem;                               // tensor_dim0_stride[31:0]
    g1[6] = 0u;
    g1[7] = 0u;
    asm volatile("tensor_load_to_lds %0, %1" :: "s"(g0), "s"(g1) : "memory");
}

// ---------------- projection GEMM: C = A[Mx1024] * W[1024x1024] + bias ------
// A: f32 (hidden) or f16 (agg). Output: either "shaped" f16 [B*H, T, D] for
// q/k/v, or f32 row-major [M, 1024] (final result).
__global__ __launch_bounds__(256)
void proj_gemm_kernel(const float*   __restrict__ Af32,
                      const _Float16* __restrict__ Af16,
                      const float*   __restrict__ W,
                      const float*   __restrict__ bias,
                      _Float16*      __restrict__ out_shaped,
                      float*         __restrict__ out_f32) {
    __shared__ _Float16 sA[128 * 64];   // [m][k]
    __shared__ _Float16 sB[128 * 64];   // [n][k] (transposed weight tile)

    const int tid = threadIdx.x;
    const int bn0 = blockIdx.x * 128;   // N offset
    const int bm0 = blockIdx.y * 128;   // M offset

    v8f acc[2][4];
#pragma unroll
    for (int i = 0; i < 2; ++i)
#pragma unroll
        for (int j = 0; j < 4; ++j) acc[i][j] = (v8f){0, 0, 0, 0, 0, 0, 0, 0};

    const int w  = tid >> 5;
    const int wm = (w & 3) * 32;        // wave M offset in block tile
    const int wn = (w >> 2) * 64;       // wave N offset in block tile

    for (int kk = 0; kk < EMB; kk += 64) {
        // stage A tile 128x64 (convert f32->f16 in the load path)
#pragma unroll 4
        for (int i = 0; i < 32; ++i) {
            const int idx = tid + 256 * i;
            const int m = idx >> 6, k = idx & 63;
            const size_t ga = (size_t)(bm0 + m) * EMB + kk + k;
            sA[idx] = Af32 ? (_Float16)Af32[ga] : Af16[ga];
        }
        // stage W tile 64x128 transposed into [n][k]
#pragma unroll 4
        for (int i = 0; i < 32; ++i) {
            const int idx = tid + 256 * i;
            const int k = idx >> 7, n = idx & 127;
            sB[n * 64 + k] = (_Float16)W[(size_t)(kk + k) * EMB + bn0 + n];
        }
        if (kk + 64 < EMB) {   // hint next K-chunk into the cache hierarchy
            const int m = tid >> 6, k = tid & 63;
            if (Af32) __builtin_prefetch(&Af32[(size_t)(bm0 + m) * EMB + kk + 64 + k], 0, 1);
            __builtin_prefetch(&W[(size_t)(kk + 64 + (tid >> 7)) * EMB + bn0 + (tid & 127)], 0, 1);
        }
        __syncthreads();

#pragma unroll
        for (int ks = 0; ks < 64; ks += 32) {
            const v16h a0 = load_a_frag(sA, 64, wm,      ks);
            const v16h a1 = load_a_frag(sA, 64, wm + 16, ks);
#pragma unroll
            for (int nt = 0; nt < 4; ++nt) {
                const v16h b = load_b_frag(sB, 64, wn + nt * 16, ks);
                acc[0][nt] = __builtin_amdgcn_wmma_f32_16x16x32_f16(
                    false, a0, false, b, (short)0, acc[0][nt], false, false);
                acc[1][nt] = __builtin_amdgcn_wmma_f32_16x16x32_f16(
                    false, a1, false, b, (short)0, acc[1][nt], false, false);
            }
        }
        __syncthreads();
    }

    // epilogue: C/D layout -> i = r + (lane>=16)*8, j = lane&15
    const int lane = tid & 31;
    const int li = (lane >> 4) << 3;
    const int lj = lane & 15;
#pragma unroll
    for (int mt = 0; mt < 2; ++mt)
#pragma unroll
        for (int nt = 0; nt < 4; ++nt)
#pragma unroll
            for (int r = 0; r < 8; ++r) {
                const int m = bm0 + wm + mt * 16 + li + r;
                const int n = bn0 + wn + nt * 16 + lj;
                const float val = acc[mt][nt][r] + bias[n];
                if (out_f32) {
                    out_f32[(size_t)m * EMB + n] = val;
                } else {
                    const int b = m >> 12, t = m & 4095;   // T = 4096
                    const int h = n >> 6,  d = n & 63;     // D = 64
                    out_shaped[((size_t)(b * HEADS + h) * T_SEQ + t) * HDIM + d] =
                        (_Float16)val;
                }
            }
}

// ---------------- fused Gram + wrapped-diagonal reduction -------------------
// mean_ac[b,tau] += (1/(H*D)) * sum_t q[bn,t,:].k[bn,t-tau,:] accumulated over
// all heads bn = b*16+h and all 128x128 (t,s) tiles.  Tiles are 16 KB
// contiguous blocks (shaped f16 layout), staged by the Tensor Data Mover.
__global__ __launch_bounds__(256)
void gram_diag_kernel(const _Float16* __restrict__ q,
                      const _Float16* __restrict__ k,
                      float* __restrict__ mean_ac) {
    __shared__ _Float16 sQ[128 * 64];
    __shared__ _Float16 sK[128 * 64];
    __shared__ float    sDiag[255];

    const int bn = blockIdx.z;
    const int t0 = blockIdx.y * 128;
    const int s0 = blockIdx.x * 128;
    const int tid = threadIdx.x;

    // wave 0 issues the two TDM copies (16 KB each); EXEC-independent DMA.
    if (tid < 32) {
        const _Float16* qsrc = q + ((size_t)bn * T_SEQ + t0) * HDIM;
        const _Float16* ksrc = k + ((size_t)bn * T_SEQ + s0) * HDIM;
        tdm_load_f16((unsigned)(uintptr_t)&sQ[0], qsrc, 128 * HDIM);
        tdm_load_f16((unsigned)(uintptr_t)&sK[0], ksrc, 128 * HDIM);
        __builtin_amdgcn_s_wait_tensorcnt(0);
    }
    if (tid < 255) sDiag[tid] = 0.0f;
    __syncthreads();

    const int w  = tid >> 5;
    const int wm = (w & 3) * 32;
    const int wn = (w >> 2) * 64;

    v8f acc[2][4];
#pragma unroll
    for (int i = 0; i < 2; ++i)
#pragma unroll
        for (int j = 0; j < 4; ++j) acc[i][j] = (v8f){0, 0, 0, 0, 0, 0, 0, 0};

#pragma unroll
    for (int ks = 0; ks < 64; ks += 32) {     // contraction over D = 64
        const v16h a0 = load_a_frag(sQ, 64, wm,      ks);
        const v16h a1 = load_a_frag(sQ, 64, wm + 16, ks);
#pragma unroll
        for (int nt = 0; nt < 4; ++nt) {
            const v16h b = load_b_frag(sK, 64, wn + nt * 16, ks);
            acc[0][nt] = __builtin_amdgcn_wmma_f32_16x16x32_f16(
                false, a0, false, b, (short)0, acc[0][nt], false, false);
            acc[1][nt] = __builtin_amdgcn_wmma_f32_16x16x32_f16(
                false, a1, false, b, (short)0, acc[1][nt], false, false);
        }
    }

    // reduce along wrapped diagonals: local tau index = (t-s) - (t0-s0) + 127
    const int lane = tid & 31;
    const int li = (lane >> 4) << 3;
    const int lj = lane & 15;
#pragma unroll
    for (int mt = 0; mt < 2; ++mt)
#pragma unroll
        for (int nt = 0; nt < 4; ++nt)
#pragma unroll
            for (int r = 0; r < 8; ++r) {
                const int ti = wm + mt * 16 + li + r;
                const int sj = wn + nt * 16 + lj;
                atomicAdd(&sDiag[ti - sj + 127], acc[mt][nt][r]);
            }
    __syncthreads();

    const int b = bn >> 4;   // bn = b*HEADS + h
    for (int idx = tid; idx < 255; idx += 256) {
        const int tau = (t0 - s0 + idx - 127) & (T_SEQ - 1);
        atomicAdd(&mean_ac[b * T_SEQ + tau], sDiag[idx] * (1.0f / 1024.0f));
    }
}

// ---------------- small helper kernels --------------------------------------
__global__ void init_mean_kernel(float* __restrict__ mean_ac) {
    mean_ac[blockIdx.x * blockDim.x + threadIdx.x] = 0.0f;
}

__global__ __launch_bounds__(256)
void topk_kernel(const float* __restrict__ mean_ac,
                 float* __restrict__ topv, int* __restrict__ topi) {
    __shared__ float sv[T_SEQ];
    __shared__ float rmax[256];
    __shared__ int   ridx[256];
    const int b = blockIdx.x, tid = threadIdx.x;

    // stage mean_ac[b] (16 KB) with async global->LDS b128 loads (ASYNCcnt)
#pragma unroll
    for (int i = 0; i < 4; ++i) {
        const int idx = tid + 256 * i;                    // 16B chunk index
        const unsigned lds_off =
            (unsigned)(uintptr_t)&sv[0] + (unsigned)idx * 16u;
        const unsigned long long ga =
            (unsigned long long)(uintptr_t)(mean_ac + (size_t)b * T_SEQ + idx * 4);
        asm volatile("global_load_async_to_lds_b128 %0, %1, off"
                     :: "v"(lds_off), "v"(ga) : "memory");
    }
    asm volatile("s_wait_asynccnt 0x0" ::: "memory");
    __syncthreads();

    for (int kk = 0; kk < TOPK; ++kk) {
        float best = -3.4e38f; int bi = 0;
        for (int i = tid; i < T_SEQ; i += 256) {
            const float v = sv[i];
            if (v > best) { best = v; bi = i; }
        }
        rmax[tid] = best; ridx[tid] = bi;
        __syncthreads();
        for (int off = 128; off > 0; off >>= 1) {
            if (tid < off && rmax[tid + off] > rmax[tid]) {
                rmax[tid] = rmax[tid + off]; ridx[tid] = ridx[tid + off];
            }
            __syncthreads();
        }
        if (tid == 0) {
            topv[b * TOPK + kk] = rmax[0];
            topi[b * TOPK + kk] = ridx[0];
            sv[ridx[0]] = -3.4e38f;
        }
        __syncthreads();
    }
}

// agg[bn,t,d] = sum_k w[bn%8,k] * v[bn,(t+delay)%T,d]; written in [B,T,E] f16.
__global__ __launch_bounds__(256)
void aggregate_kernel(const _Float16* __restrict__ v,
                      const float* __restrict__ topv,
                      const int*   __restrict__ topi,
                      _Float16* __restrict__ agg) {
    __shared__ float sw[TOPK];
    __shared__ int   sd[TOPK];
    const int bn  = blockIdx.y;
    const int tid = threadIdx.x;
    const int bw  = bn & 7;            // reference's bn % B quirk
    if (tid < TOPK) { sw[tid] = topv[bw * TOPK + tid]; sd[tid] = topi[bw * TOPK + tid]; }
    __syncthreads();

    const int d  = tid & 63;
    const int tt = blockIdx.x * 4 + (tid >> 6);
    const _Float16* vb = v + (size_t)bn * T_SEQ * HDIM;
    float acc = 0.0f;
#pragma unroll
    for (int kk = 0; kk < TOPK; ++kk) {
        const int st = (tt + sd[kk]) & (T_SEQ - 1);
        acc += sw[kk] * (float)vb[(size_t)st * HDIM + d];
    }
    const int b = bn >> 4, h = bn & 15;
    agg[((size_t)b * T_SEQ + tt) * EMB + h * HDIM + d] = (_Float16)acc;
}

// ---------------- launcher --------------------------------------------------
extern "C" void kernel_launch(void* const* d_in, const int* in_sizes, int n_in,
                              void* d_out, int out_size, void* d_ws, size_t ws_size,
                              hipStream_t stream) {
    (void)in_sizes; (void)n_in; (void)out_size; (void)ws_size;
    const float* X  = (const float*)d_in[0];
    const float* Wq = (const float*)d_in[1]; const float* bq = (const float*)d_in[2];
    const float* Wk = (const float*)d_in[3]; const float* bk = (const float*)d_in[4];
    const float* Wv = (const float*)d_in[5]; const float* bv = (const float*)d_in[6];
    const float* Wo = (const float*)d_in[7]; const float* bo = (const float*)d_in[8];
    float* out = (float*)d_out;

    // workspace layout (~269 MB): q,k,v,agg f16 + mean_ac + topk
    const size_t QKV = (size_t)BNHEADS * T_SEQ * HDIM;     // 33_554_432 elems
    _Float16* q   = (_Float16*)d_ws;
    _Float16* k   = q + QKV;
    _Float16* v   = k + QKV;
    _Float16* agg = v + QKV;
    float* mean_ac = (float*)(agg + QKV);
    float* topv    = mean_ac + (size_t)BATCH * T_SEQ;
    int*   topi    = (int*)(topv + BATCH * TOPK);

    const dim3 gemm_grid(EMB / 128, MROWS / 128);          // (8, 256)

    proj_gemm_kernel<<<gemm_grid, 256, 0, stream>>>(X, nullptr, Wq, bq, q, nullptr);
    proj_gemm_kernel<<<gemm_grid, 256, 0, stream>>>(X, nullptr, Wk, bk, k, nullptr);
    proj_gemm_kernel<<<gemm_grid, 256, 0, stream>>>(X, nullptr, Wv, bv, v, nullptr);

    init_mean_kernel<<<(BATCH * T_SEQ) / 1024, 1024, 0, stream>>>(mean_ac);
    gram_diag_kernel<<<dim3(T_SEQ / 128, T_SEQ / 128, BNHEADS), 256, 0, stream>>>(q, k, mean_ac);

    topk_kernel<<<BATCH, 256, 0, stream>>>(mean_ac, topv, topi);
    aggregate_kernel<<<dim3(T_SEQ / 4, BNHEADS), 256, 0, stream>>>(v, topv, topi, agg);

    proj_gemm_kernel<<<gemm_grid, 256, 0, stream>>>(nullptr, agg, Wo, bo, nullptr, out);
}
